// QKVLinear_48730698940882
// MI455X (gfx1250) — matrix operand order repaired and checked by reading, not compiled, not run on hardware
//
#include <hip/hip_runtime.h>

// CDNA5 / gfx1250 QKV projection: C = q @ W^T + b  (fp32 in/out)
// Pass 1: q,W fp32 -> bf16 hi + residual-lo planes in d_ws (converted ONCE).
// Pass 2: GEMM, 3x V_WMMA_F32_16X16X32_BF16 per tile per K-step
//         (hi*hi + lo*hi + hi*lo ~ fp32). Hot loop moves bf16 tiles with
//         GLOBAL_LOAD_ASYNC_TO_LDS_B128 (ASYNCcnt) -> ds_load_b128 -> WMMA.
// Double-buffered LDS, NT output stores keep operand planes resident in L2.

typedef __attribute__((ext_vector_type(16))) __bf16 v16bf;
typedef __attribute__((ext_vector_type(8)))  __bf16 v8bf;
typedef __attribute__((ext_vector_type(4)))  __bf16 v4bf;
typedef __attribute__((ext_vector_type(8)))  float  v8f;
typedef __attribute__((ext_vector_type(4)))  float  v4f;

#define M_TOTAL 16384
#define N_TOTAL 3072
#define K_TOTAL 1024
#define BM 64
#define BN 128
#define BK 32
#define NK (K_TOTAL / BK)
#define STR 40   // bf16 elems per LDS row (80 B: 16B-aligned chunks, bank-spread)

static __device__ __forceinline__ v16bf cat8(v8bf a, v8bf b) {
    return __builtin_shufflevector(a, b, 0, 1, 2, 3, 4, 5, 6, 7,
                                         8, 9, 10, 11, 12, 13, 14, 15);
}

// ---- CDNA5 async global->LDS copy (VGLOBAL op 95..98, tracked by ASYNCcnt) ----
static __device__ __forceinline__ void async_copy_b128(unsigned lds_off, const void* gptr) {
    asm volatile("global_load_async_to_lds_b128 %0, %1, off"
                 :
                 : "v"(lds_off), "v"(gptr)
                 : "memory");
}

static __device__ __forceinline__ void wait_async_all() {
#if __has_builtin(__builtin_amdgcn_s_wait_asynccnt)
    __builtin_amdgcn_s_wait_asynccnt(0);
#else
    asm volatile("s_wait_asynccnt 0x0" ::: "memory");
#endif
}

static __device__ __forceinline__ unsigned lds_addr32(const void* p) {
    return (unsigned)(uintptr_t)p;   // LDS aperture: addr[31:0] is the LDS byte address
}

// ---------------------------------------------------------------------------
// Pass 1: fp32 -> bf16 hi/lo planes (each element converted exactly once)
// ---------------------------------------------------------------------------
#define NQ4 ((M_TOTAL * (size_t)K_TOTAL) / 4)   // 4194304 float4s of q
#define NW4 ((N_TOTAL * (size_t)K_TOTAL) / 4)   //  786432 float4s of W

__launch_bounds__(256)
__global__ void cvt_kernel(const float* __restrict__ q,
                           const float* __restrict__ w,
                           __bf16* __restrict__ qh, __bf16* __restrict__ ql,
                           __bf16* __restrict__ wh, __bf16* __restrict__ wl) {
    size_t i = (size_t)blockIdx.x * 256 + threadIdx.x;
    const v4f* src;
    v4bf* dh;
    v4bf* dl;
    size_t j;
    if (i < NQ4) {
        src = (const v4f*)q; dh = (v4bf*)qh; dl = (v4bf*)ql; j = i;
    } else {
        src = (const v4f*)w; dh = (v4bf*)wh; dl = (v4bf*)wl; j = i - NQ4;
        if (j >= NW4) return;
    }
    v4f d = src[j];
    v4bf h, l;
    #pragma unroll
    for (int k = 0; k < 4; ++k) {
        __bf16 hh = (__bf16)d[k];
        h[k] = hh;
        l[k] = (__bf16)(d[k] - (float)hh);
    }
    dh[j] = h;
    dl[j] = l;
}

// ---------------------------------------------------------------------------
// Pass 2: bf16-plane GEMM. Wave tile 32x32 (2 M-tiles x 2 N-tiles).
// ---------------------------------------------------------------------------
__launch_bounds__(256)
__global__ void qkv_wmma_kernel(const __bf16* __restrict__ qh_g,
                                const __bf16* __restrict__ ql_g,
                                const __bf16* __restrict__ wh_g,
                                const __bf16* __restrict__ wl_g,
                                const float* __restrict__ bias,
                                float* __restrict__ out) {
    __shared__ __bf16 qh[2][BM * STR];
    __shared__ __bf16 ql[2][BM * STR];
    __shared__ __bf16 wh[2][BN * STR];
    __shared__ __bf16 wl[2][BN * STR];

    const int tid   = threadIdx.x;
    const int lane  = tid & 31;
    const int wave  = tid >> 5;
    const int wm    = wave & 1;    // 32-row strip inside BM
    const int wn    = wave >> 1;   // 32-col strip inside BN

    const int blockN = blockIdx.x * BN;
    const int blockM = blockIdx.y * BM;

    const int l15   = lane & 15;
    const int lhalf = lane >> 4;
    const int sA    = lhalf * 8;    // A-frag K sub-offset
    const int oB    = lhalf * 8;    // B-frag K sub-offset

    // staging coords: thread -> row rq, 8 bf16 at col c8 (q: rows 0..63; W: +64 for rep 1)
    const int rq = tid >> 2;
    const int c8 = (tid & 3) * 8;
    const __bf16* qhp = qh_g + (size_t)(blockM + rq) * K_TOTAL + c8;
    const __bf16* qlp = ql_g + (size_t)(blockM + rq) * K_TOTAL + c8;
    const __bf16* whp = wh_g + (size_t)(blockN + rq) * K_TOTAL + c8;
    const __bf16* wlp = wl_g + (size_t)(blockN + rq) * K_TOTAL + c8;
    const size_t wstep = (size_t)64 * K_TOTAL;

    // per-thread LDS destinations (byte addresses), both buffers
    const int idx = rq * STR + c8;
    unsigned l_qh[2], l_ql[2], l_wh0[2], l_wl0[2], l_wh1[2], l_wl1[2];
    #pragma unroll
    for (int b = 0; b < 2; ++b) {
        l_qh[b]  = lds_addr32(&qh[b][idx]);
        l_ql[b]  = lds_addr32(&ql[b][idx]);
        l_wh0[b] = lds_addr32(&wh[b][idx]);
        l_wl0[b] = lds_addr32(&wl[b][idx]);
        l_wh1[b] = lds_addr32(&wh[b][idx + 64 * STR]);
        l_wl1[b] = lds_addr32(&wl[b][idx + 64 * STR]);
    }

    v8f acc[2][2];
    #pragma unroll
    for (int mi = 0; mi < 2; ++mi)
        #pragma unroll
        for (int ni = 0; ni < 2; ++ni) acc[mi][ni] = (v8f){};

    // ---- prologue: async-stage K-step 0 into buffer 0 ----
    async_copy_b128(l_qh[0], qhp);
    async_copy_b128(l_ql[0], qlp);
    async_copy_b128(l_wh0[0], whp);
    async_copy_b128(l_wl0[0], wlp);
    async_copy_b128(l_wh1[0], whp + wstep);
    async_copy_b128(l_wl1[0], wlp + wstep);
    wait_async_all();
    __syncthreads();

    #pragma unroll 2
    for (int i = 0; i < NK; ++i) {
        const int buf = i & 1;
        const int nxt = buf ^ 1;

        // ---- issue next K-step's async copies (overlap with WMMAs) ----
        if (i + 1 < NK) {
            const int kb = (i + 1) * BK;
            async_copy_b128(l_qh[nxt], qhp + kb);
            async_copy_b128(l_ql[nxt], qlp + kb);
            async_copy_b128(l_wh0[nxt], whp + kb);
            async_copy_b128(l_wl0[nxt], wlp + kb);
            async_copy_b128(l_wh1[nxt], whp + kb + wstep);
            async_copy_b128(l_wl1[nxt], wlp + kb + wstep);
        }

        // ---- compute from current buffer ----
        const __bf16* qh_c = qh[buf];
        const __bf16* ql_c = ql[buf];
        const __bf16* wh_c = wh[buf];
        const __bf16* wl_c = wl[buf];

        v16bf a_hi[2], a_lo[2], b_hi[2], b_lo[2];
        #pragma unroll
        for (int mi = 0; mi < 2; ++mi) {
            const int arow = (wm * 32 + mi * 16 + l15) * STR;
            a_hi[mi] = cat8(*(const v8bf*)&qh_c[arow + sA],
                            *(const v8bf*)&qh_c[arow + 16 + sA]);
            a_lo[mi] = cat8(*(const v8bf*)&ql_c[arow + sA],
                            *(const v8bf*)&ql_c[arow + 16 + sA]);
        }
        #pragma unroll
        for (int ni = 0; ni < 2; ++ni) {
            const int brow = (wn * 32 + ni * 16 + l15) * STR + oB;
            b_hi[ni] = cat8(*(const v8bf*)&wh_c[brow],
                            *(const v8bf*)&wh_c[brow + 16]);
            b_lo[ni] = cat8(*(const v8bf*)&wl_c[brow],
                            *(const v8bf*)&wl_c[brow + 16]);
        }

        #pragma unroll
        for (int mi = 0; mi < 2; ++mi) {
            #pragma unroll
            for (int ni = 0; ni < 2; ++ni) {
                acc[mi][ni] = __builtin_amdgcn_wmma_f32_16x16x32_bf16(
                    false, a_hi[mi], false, b_hi[ni], (short)0, acc[mi][ni], false, false);
                acc[mi][ni] = __builtin_amdgcn_wmma_f32_16x16x32_bf16(
                    false, a_lo[mi], false, b_hi[ni], (short)0, acc[mi][ni], false, false);
                acc[mi][ni] = __builtin_amdgcn_wmma_f32_16x16x32_bf16(
                    false, a_hi[mi], false, b_lo[ni], (short)0, acc[mi][ni], false, false);
            }
        }

        // ---- drain async copies, then make next buffer visible to all waves ----
        if (i + 1 < NK) wait_async_all();
        __syncthreads();
    }

    // ---- epilogue: bias add + non-temporal store (keep planes resident in L2) ----
    #pragma unroll
    for (int mi = 0; mi < 2; ++mi) {
        #pragma unroll
        for (int ni = 0; ni < 2; ++ni) {
            int n = blockN + wn * 32 + ni * 16 + l15;
            float bv = bias[n];
            #pragma unroll
            for (int r = 0; r < 8; ++r) {
                int m = blockM + wm * 32 + mi * 16 + r + lhalf * 8;
                __builtin_nontemporal_store(acc[mi][ni][r] + bv,
                                            &out[(size_t)m * N_TOTAL + n]);
            }
        }
    }
}

// ---------------------------------------------------------------------------
// Fallback (no scratch): convert in staging
// ---------------------------------------------------------------------------
static __device__ __forceinline__ void cvt_store_fb(v4f d, __bf16* hi, __bf16* lo) {
    v4bf h, l;
    #pragma unroll
    for (int j = 0; j < 4; ++j) {
        __bf16 hh = (__bf16)d[j];
        h[j] = hh;
        l[j] = (__bf16)(d[j] - (float)hh);
    }
    *(v4bf*)hi = h;
    *(v4bf*)lo = l;
}

__launch_bounds__(256)
__global__ void qkv_wmma_fallback(const float* __restrict__ q,
                                  const float* __restrict__ w,
                                  const float* __restrict__ bias,
                                  float* __restrict__ out) {
    __shared__ __bf16 qh[2][BM * STR];
    __shared__ __bf16 ql[2][BM * STR];
    __shared__ __bf16 wh[2][BN * STR];
    __shared__ __bf16 wl[2][BN * STR];

    const int tid   = threadIdx.x;
    const int lane  = tid & 31;
    const int wave  = tid >> 5;
    const int wm    = wave & 3;
    const int wn    = wave >> 2;
    const int blockN = blockIdx.x * BN;
    const int blockM = blockIdx.y * BM;
    const int l15   = lane & 15;
    const int lhalf = lane >> 4;
    const int sA    = lhalf * 8;
    const int oB    = lhalf * 8;

    const int r0 = tid >> 3;
    const int c4 = (tid & 7) * 4;
    const float* qp = q + (size_t)(blockM + r0) * K_TOTAL + c4;
    const float* wp = w + (size_t)(blockN + r0) * K_TOTAL + c4;

    v8f acc[4];
    #pragma unroll
    for (int t = 0; t < 4; ++t) acc[t] = (v8f){};

    v4f qreg[2], wreg[4];
    #pragma unroll
    for (int rep = 0; rep < 2; ++rep)
        qreg[rep] = *(const v4f*)(qp + (size_t)rep * 32 * K_TOTAL);
    #pragma unroll
    for (int rep = 0; rep < 4; ++rep)
        wreg[rep] = *(const v4f*)(wp + (size_t)rep * 32 * K_TOTAL);
    #pragma unroll
    for (int rep = 0; rep < 2; ++rep) {
        int idx = (r0 + rep * 32) * STR + c4;
        cvt_store_fb(qreg[rep], &qh[0][idx], &ql[0][idx]);
    }
    #pragma unroll
    for (int rep = 0; rep < 4; ++rep) {
        int idx = (r0 + rep * 32) * STR + c4;
        cvt_store_fb(wreg[rep], &wh[0][idx], &wl[0][idx]);
    }
    __syncthreads();

    #pragma unroll 2
    for (int i = 0; i < NK; ++i) {
        const int buf = i & 1;
        const int nxt = buf ^ 1;
        if (i + 1 < NK) {
            const int kb = (i + 1) * BK;
            #pragma unroll
            for (int rep = 0; rep < 2; ++rep)
                qreg[rep] = *(const v4f*)(qp + kb + (size_t)rep * 32 * K_TOTAL);
            #pragma unroll
            for (int rep = 0; rep < 4; ++rep)
                wreg[rep] = *(const v4f*)(wp + kb + (size_t)rep * 32 * K_TOTAL);
        }
        const __bf16* qh_c = qh[buf];
        const __bf16* ql_c = ql[buf];
        const __bf16* wh_c = wh[buf];
        const __bf16* wl_c = wl[buf];
        const int arow = (wm * 16 + l15) * STR;
        v16bf a_hi = cat8(*(const v8bf*)&qh_c[arow + sA],
                          *(const v8bf*)&qh_c[arow + 16 + sA]);
        v16bf a_lo = cat8(*(const v8bf*)&ql_c[arow + sA],
                          *(const v8bf*)&ql_c[arow + 16 + sA]);
        #pragma unroll
        for (int t = 0; t < 4; ++t) {
            const int brow = (wn * 64 + t * 16 + l15) * STR + oB;
            v16bf b_hi = cat8(*(const v8bf*)&wh_c[brow],
                              *(const v8bf*)&wh_c[brow + 16]);
            v16bf b_lo = cat8(*(const v8bf*)&wl_c[brow],
                              *(const v8bf*)&wl_c[brow + 16]);
            acc[t] = __builtin_amdgcn_wmma_f32_16x16x32_bf16(
                false, a_hi, false, b_hi, (short)0, acc[t], false, false);
            acc[t] = __builtin_amdgcn_wmma_f32_16x16x32_bf16(
                false, a_lo, false, b_hi, (short)0, acc[t], false, false);
            acc[t] = __builtin_amdgcn_wmma_f32_16x16x32_bf16(
                false, a_hi, false, b_lo, (short)0, acc[t], false, false);
        }
        if (i + 1 < NK) {
            #pragma unroll
            for (int rep = 0; rep < 2; ++rep) {
                int idx2 = (r0 + rep * 32) * STR + c4;
                cvt_store_fb(qreg[rep], &qh[nxt][idx2], &ql[nxt][idx2]);
            }
            #pragma unroll
            for (int rep = 0; rep < 4; ++rep) {
                int idx2 = (r0 + rep * 32) * STR + c4;
                cvt_store_fb(wreg[rep], &wh[nxt][idx2], &wl[nxt][idx2]);
            }
        }
        __syncthreads();
    }

    #pragma unroll
    for (int t = 0; t < 4; ++t) {
        int n = blockN + wn * 64 + t * 16 + l15;
        float bv = bias[n];
        #pragma unroll
        for (int r = 0; r < 8; ++r) {
            int m = blockM + wm * 16 + r + lhalf * 8;
            __builtin_nontemporal_store(acc[t][r] + bv,
                                        &out[(size_t)m * N_TOTAL + n]);
        }
    }
}

// ---------------------------------------------------------------------------
extern "C" void kernel_launch(void* const* d_in, const int* in_sizes, int n_in,
                              void* d_out, int out_size, void* d_ws, size_t ws_size,
                              hipStream_t stream) {
    const float* q    = (const float*)d_in[0];   // [16384, 1024]
    const float* wqkv = (const float*)d_in[1];   // [3072, 1024]
    const float* bqkv = (const float*)d_in[2];   // [3072]
    float* out = (float*)d_out;                  // [16384, 3072]

    const size_t MSZ = (size_t)M_TOTAL * K_TOTAL * sizeof(__bf16); // 33554432
    const size_t WSZ = (size_t)N_TOTAL * K_TOTAL * sizeof(__bf16); //  6291456
    const size_t REQ = 2 * MSZ + 2 * WSZ;                          // ~76 MiB

    dim3 gemm_grid(N_TOTAL / BN, M_TOTAL / BM);  // 24 x 256
    dim3 block(256);

    if (ws_size >= REQ) {
        __bf16* qh = (__bf16*)d_ws;
        __bf16* ql = (__bf16*)((char*)d_ws + MSZ);
        __bf16* wh = (__bf16*)((char*)d_ws + 2 * MSZ);
        __bf16* wl = (__bf16*)((char*)d_ws + 2 * MSZ + WSZ);

        const size_t total4 = NQ4 + NW4;                 // 4980736 float4s
        dim3 cvt_grid((unsigned)((total4 + 255) / 256)); // 19456 blocks
        cvt_kernel<<<cvt_grid, block, 0, stream>>>(q, wqkv, qh, ql, wh, wl);
        qkv_wmma_kernel<<<gemm_grid, block, 0, stream>>>(qh, ql, wh, wl, bqkv, out);
    } else {
        qkv_wmma_fallback<<<gemm_grid, block, 0, stream>>>(q, wqkv, bqkv, out);
    }
}
